// XrayForwardProjection_32469952758432
// MI455X (gfx1250) — matrix operand order repaired and checked by reading, not compiled, not run on hardware
//
#include <hip/hip_runtime.h>
#include <hip/hip_bf16.h>

// X-ray parallel-beam forward projection for MI455X (gfx1250, wave32).
//
//   Kernel 1: vol (Z,Y,X f32) -> volP (Y,X,Z f16-pair): element (y,x,z) is
//             the 4-byte pair {v[y,x,z], v[y,x+1,z]} (x clamped). z is the
//             contiguous axis; 64 MiB stays resident in the 192 MB L2.
//   Kernel 2: per wave: one (angle, u, 128-z block). Contraction over
//             k = (8 t-steps x 4 bilinear corners) = K=32 runs on the
//             matrix pipe: V_WMMA_F32_16X16X32_F16, A = weight rows
//             (z-independent, replicated), B = gathered volume columns.
//             Each B VGPR = one b32 gather (x-pair layout => the two
//             16-bit halves are exactly K=2v,2v+1); 8 z-tiles share one
//             A matrix and one 8-entry offset set so the per-chunk
//             weight/address VALU is amortized over 8 WMMAs. Gathers are
//             GVS-addressed (SGPR base + 32-bit lane offset + imm z-tile
//             offset), issued q-major so each WMMA waits only on its own
//             group. f32 accumulators.
//
// NOTE: requires ws_size >= 256*256*256*4 bytes (64 MiB) for volP.

typedef __attribute__((ext_vector_type(16))) _Float16 v16h;
typedef __attribute__((ext_vector_type(2)))  _Float16 h2;
typedef __attribute__((ext_vector_type(8)))  float    v8f;

#define NA   180
#define DIMN 256     // X = Y = Z = det_u = 256
#define NT   368
#define TCH  46      // 368 / 8 t-chunks
#define NQT  8       // z-tiles per wave (128 z)

// ---------- build pair volume: (Z,Y,X) f32 -> (Y,X,Z) {f16,f16} ----------
__global__ __launch_bounds__(256) void xray_pairvol_kernel(
    const float* __restrict__ vol, h2* __restrict__ volP) {
  __shared__ float tile[32][33];         // [z_local][x_local], 0..32 in x
  const int tx = threadIdx.x;            // 0..31
  const int ty = threadIdx.y;            // 0..7
  const int x0 = blockIdx.x * 32;
  const int z0 = blockIdx.y * 32;
  const int y  = blockIdx.z;
  #pragma unroll
  for (int r = 0; r < 4; ++r) {
    int zz = z0 + ty + 8 * r;
    tile[ty + 8 * r][tx] = vol[((size_t)zz * DIMN + y) * DIMN + (x0 + tx)];
  }
  if (ty == 0) {                         // extra column for the x+1 neighbor
    int xe = min(x0 + 32, DIMN - 1);
    tile[tx][32] = vol[((size_t)(z0 + tx) * DIMN + y) * DIMN + xe];
  }
  __syncthreads();
  #pragma unroll
  for (int r = 0; r < 4; ++r) {
    int xl = ty + 8 * r;
    h2 pr;
    pr.x = (_Float16)tile[tx][xl];
    pr.y = (_Float16)tile[tx][xl + 1];
    volP[((size_t)y * DIMN + (x0 + xl)) * DIMN + (z0 + tx)] = pr;
  }
}

// ---------- per-(t) sample evaluation ----------
struct EvalR {
  float wx, wy, inb;
  int x0i, y0i, y1i;
};

__device__ __forceinline__ EvalR evalT(float uu, float cth, float sth, float tglob) {
  const float tv = tglob - (float)((NT - 1) * 0.5);     // 183.5
  const float xs = fmaf(uu, cth, fmaf(-tv, sth, 127.5f));
  const float ys = fmaf(uu, sth, fmaf( tv, cth, 127.5f));
  const float x0 = floorf(xs);
  const float y0 = floorf(ys);
  EvalR e;
  e.wx  = xs - x0;
  e.wy  = ys - y0;
  e.inb = (xs >= 0.f && xs <= 255.f && ys >= 0.f && ys <= 255.f) ? 1.f : 0.f;
  int xi = (int)x0, yi = (int)y0;
  e.x0i = min(max(xi, 0), DIMN - 1);     // x+1 clamp is baked into volP
  e.y0i = min(max(yi, 0), DIMN - 1);
  e.y1i = min(e.y0i + 1, DIMN - 1);
  return e;
}

union Bmat { v16h v; h2 g[8]; };

// ---------- projection: WMMA over (t,corner) contraction ----------
__global__ __launch_bounds__(32) void xray_project_kernel(
    const h2* __restrict__ volP,         // (Y, X, Z) f16-pairs
    const float* __restrict__ angles,    // (NA)
    float* __restrict__ out) {           // (NA, Z, det_u)
  const int lane = threadIdx.x;          // 0..31
  const int half = lane >> 4;            // K-block selector
  const int a    = blockIdx.x;           // angle
  const int zb   = blockIdx.y;           // 0..1  (128 z each)
  const int u    = blockIdx.z;           // 0..255

  const float theta = angles[a];
  const float cth = cosf(theta);
  const float sth = sinf(theta);
  const float uu  = (float)u - (float)((DIMN - 1) * 0.5);   // u - 127.5
  const unsigned zoff = (unsigned)(zb * 128 + (lane & 15)); // lane z offset

  v8f acc[NQT];
  #pragma unroll
  for (int q = 0; q < NQT; ++q) acc[q] = (v8f){};

  #pragma unroll 1
  for (int ch = 0; ch < TCH; ++ch) {
    const float tbase = (float)(ch * 8);

    // -- Phase 1: 8 gather offsets (32-bit, uniform base) ---------------
    // B VGPR pair r = 2*jj + row: K=(4jj+2row, 4jj+2row+1) = x-pair
    unsigned off[8];
    #pragma unroll
    for (int jj = 0; jj < 4; ++jj) {
      const int j = half * 4 + jj;                   // t-offset of K group
      EvalR e = evalT(uu, cth, sth, tbase + (float)j);
      const unsigned col = (unsigned)(e.x0i << 8) + zoff;
      off[2 * jj + 0] = (unsigned)(e.y0i << 16) + col;   // row y0
      off[2 * jj + 1] = (unsigned)(e.y1i << 16) + col;   // row y1
    }

    // -- Phase 2: 64 b32 gathers, one per B VGPR, q-major order ---------
    Bmat b[NQT];
    #pragma unroll
    for (int q = 0; q < NQT; ++q) {
      #pragma unroll
      for (int r = 0; r < 8; ++r) b[q].g[r] = volP[off[r] + 16 * q];
    }

    // -- Phase 3: A weights (independent VALU in the load shadow) -------
    // A layout: half 0 holds K 0-15; half 1 holds K 16-31
    v16h aw;
    #pragma unroll
    for (int g = 0; g < 4; ++g) {
      const int j = g + half * 4;
      EvalR e = evalT(uu, cth, sth, tbase + (float)j);
      const float omx = 1.f - e.wx;
      const float omy = 1.f - e.wy;
      aw[g * 4 + 0] = (_Float16)(omx  * omy  * e.inb);
      aw[g * 4 + 1] = (_Float16)(e.wx * omy  * e.inb);
      aw[g * 4 + 2] = (_Float16)(omx  * e.wy * e.inb);
      aw[g * 4 + 3] = (_Float16)(e.wx * e.wy * e.inb);
    }

    // -- Phase 4: matrix-pipe accumulation: D += A(16x32) @ B(32x16) ----
    #pragma unroll
    for (int q = 0; q < NQT; ++q) {
      acc[q] = __builtin_amdgcn_wmma_f32_16x16x32_f16(false, aw, false, b[q].v,
                                                      (short)0, acc[q], false, false);
    }
  }

  // D row 0 lives in lanes 0-15, first accumulator register (n = lane).
  if (lane < 16) {
    float* o = out + ((size_t)a * DIMN) * DIMN + u;  // out[a][z][u]
    #pragma unroll
    for (int q = 0; q < NQT; ++q) {
      o[(size_t)(zb * 128 + 16 * q + lane) * DIMN] = acc[q][0];
    }
  }
}

extern "C" void kernel_launch(void* const* d_in, const int* in_sizes, int n_in,
                              void* d_out, int out_size, void* d_ws, size_t ws_size,
                              hipStream_t stream) {
  const float* vol    = (const float*)d_in[0];   // (1,1,256,256,256) f32
  const float* angles = (const float*)d_in[1];   // (180) f32
  float*       out    = (float*)d_out;           // (1,1,180,256,256) f32
  h2*          volP   = (h2*)d_ws;               // needs 64 MiB

  dim3 tb(32, 8, 1);
  dim3 tg(DIMN / 32, DIMN / 32, DIMN);
  xray_pairvol_kernel<<<tg, tb, 0, stream>>>(vol, volP);

  dim3 pb(32, 1, 1);
  dim3 pg(NA, 2, DIMN);
  xray_project_kernel<<<pg, pb, 0, stream>>>(volP, angles, out);
}
